// wiseNet_15539191677379
// MI455X (gfx1250) — matrix operand order repaired and checked by reading, not compiled, not run on hardware
//
#include <hip/hip_runtime.h>
#include <hip/hip_bf16.h>

typedef __attribute__((ext_vector_type(16))) _Float16     v16h;
typedef __attribute__((ext_vector_type(8)))  float        v8f;
typedef __attribute__((ext_vector_type(4)))  unsigned int u32x4;
typedef __attribute__((ext_vector_type(8)))  unsigned int u32x8;

// ===================== small generic kernels =====================

__global__ void cast_f32_f16_kernel(const float* __restrict__ x,
                                    _Float16* __restrict__ y, long n) {
  long i = (long)blockIdx.x * blockDim.x + threadIdx.x;
  if (i < n) y[i] = (_Float16)x[i];
}

// scale = g / sqrt(v+eps); shift = (conv_bias - m)*scale + b
__global__ void bn_prep_kernel(const float* __restrict__ g, const float* __restrict__ bb,
                               const float* __restrict__ m, const float* __restrict__ v,
                               const float* __restrict__ cb,
                               float* __restrict__ scale, float* __restrict__ shift, int C) {
  int c = blockIdx.x * blockDim.x + threadIdx.x;
  if (c >= C) return;
  float sc = g[c] * rsqrtf(v[c] + 1e-5f);
  scale[c] = sc;
  shift[c] = (cb[c] - m[c]) * sc + bb[c];
}

__global__ void maxpool3d_kernel(const _Float16* __restrict__ x, _Float16* __restrict__ y,
                                 long total, int ID, int OD, int k, int s) {
  long i = (long)blockIdx.x * blockDim.x + threadIdx.x;
  if (i >= total) return;
  int ox = (int)(i % OD); long t = i / OD;
  int oy = (int)(t % OD); t /= OD;
  int oz = (int)(t % OD); long nc = t / OD;
  const _Float16* xp = x + nc * (long)ID * ID * ID;
  float best = -3.0e38f;
  for (int dz = 0; dz < k; ++dz)
    for (int dy = 0; dy < k; ++dy)
      for (int dx = 0; dx < k; ++dx) {
        int iz = oz * s + dz, iy = oy * s + dy, ix = ox * s + dx;
        float v = (float)xp[((long)iz * ID + iy) * ID + ix];
        best = v > best ? v : best;
      }
  y[i] = (_Float16)best;
}

__global__ void meanpool_kernel(const _Float16* __restrict__ x, float* __restrict__ y,
                                long NC, int V) {
  long i = (long)blockIdx.x * blockDim.x + threadIdx.x;
  if (i >= NC) return;
  const _Float16* xp = x + i * (long)V;
  float s = 0.f;
  for (int j = 0; j < V; ++j) s += (float)xp[j];
  y[i] = s / (float)V;
}

// y[n,o] = b[o] + sum_j x[n,j] * w[o,j]
__global__ void linear_kernel(const float* __restrict__ x, const float* __restrict__ w,
                              const float* __restrict__ b, float* __restrict__ y,
                              int N, int IN, int OUT) {
  int i = blockIdx.x * blockDim.x + threadIdx.x;
  if (i >= N * OUT) return;
  int n = i / OUT, o = i % OUT;
  const float* xp = x + (long)n * IN;
  const float* wp = w + (long)o * IN;
  float s = b[o];
  for (int j = 0; j < IN; ++j) s += xp[j] * wp[j];
  y[i] = s;
}

// ===================== implicit-GEMM conv3d: TDM weight staging + WMMA =====================
// x: [N, CIN, ID^3] f16, wgt: [COUT, CIN*KS^3] f16, y: [N, COUT, OD^3] f16
// GEMM: M = OD^3 (output voxels), N = COUT, K = CIN*KS^3.
// Per block: TDM DMAs the 16xKTOT f16 weight tile for blockIdx.y into LDS once;
// 4 waves each compute one 16x16 D-tile reading B-fragments from LDS.
template <int CIN, int COUT, int KS, int STR, int PAD, int DIL, int ID, int OD>
__global__ __launch_bounds__(128)
void conv3d_wmma_kernel(const _Float16* __restrict__ x,
                        const _Float16* __restrict__ wgt,
                        const float* __restrict__ scale,
                        const float* __restrict__ shift,
                        _Float16* __restrict__ y, int mtiles) {
  constexpr int KTOT = CIN * KS * KS * KS;
  constexpr int KCH  = (KTOT + 31) / 32;
  constexpr int M    = OD * OD * OD;

  __shared__ _Float16 lds_w[16 * KTOT];   // compact [16][KTOT] tile (TDM layout)

  const int lane = threadIdx.x & 31;
  const int wv   = threadIdx.x >> 5;
  const int mt   = blockIdx.x * 4 + wv;
  const int nt   = blockIdx.y;
  const int n    = blockIdx.z;

  // ---- Tensor Data Mover: stage weight tile [nt*16 .. nt*16+15][0..KTOT) -> LDS ----
  if (wv == 0) {
    unsigned lds_off = (unsigned)(size_t)(&lds_w[0]);   // low 32 bits of LDS aperture addr
    unsigned long long ga =
        (unsigned long long)(size_t)(wgt + (size_t)nt * 16 * KTOT);
    u32x4 g0; u32x8 g1;
    // D# group0: count=1, lds_addr, 57-bit global addr, type=2 ("image")
    g0[0] = 1u;
    g0[1] = lds_off;
    g0[2] = (unsigned)(ga & 0xFFFFFFFFull);
    g0[3] = (unsigned)((ga >> 32) & 0x01FFFFFFull) | 0x80000000u;
    // D# group1: data_size=1 (2 bytes); tensor_dim0=KTOT; tensor_dim1=COUT;
    // tile_dim0=KTOT; tile_dim1=16; tile_dim2=0; dim0_stride=KTOT; dim1_stride=0
    g1[0] = 0x00010000u;
    g1[1] = ((unsigned)KTOT & 0xFFFFu) << 16;
    g1[2] = (((unsigned)KTOT >> 16) & 0xFFFFu) | (((unsigned)COUT & 0xFFFFu) << 16);
    g1[3] = (((unsigned)COUT >> 16) & 0xFFFFu) | (((unsigned)KTOT & 0xFFFFu) << 16);
    g1[4] = 16u;
    g1[5] = (unsigned)KTOT;
    g1[6] = 0u;
    g1[7] = 0u;
    asm volatile("tensor_load_to_lds %0, %1" ::"s"(g0), "s"(g1) : "memory");
    __builtin_amdgcn_s_wait_tensorcnt(0);
  }
  __syncthreads();   // publish LDS tile to all 4 waves (no early return before this!)

  if (mt < mtiles) {                        // uniform per wave
    const int mrow16 = lane & 15;           // A row / B-D column selector
    const int m0     = mt * 16 + mrow16;    // this lane's A row (M index)
    const bool mvalid = (m0 < M);
    int oz = 0, oy = 0, ox = 0;
    {
      int mm = mvalid ? m0 : 0;
      oz = mm / (OD * OD);
      int r = mm - oz * OD * OD;
      oy = r / OD;
      ox = r - oy * OD;
    }
    const _Float16* __restrict__ xin = x + (size_t)n * CIN * ID * ID * ID;
    const int ncol   = nt * 16 + mrow16;    // this lane's B/D column (channel)
    const int kbaseA = (lane < 16) ? 0 : 8;   // ISA 16-bit A 16x32 layout
    const int kbaseB = (lane < 16) ? 0 : 8;   // ISA 16-bit B 32x16 layout

    v8f acc = {};
#pragma unroll 1
    for (int kc = 0; kc < KCH; ++kc) {
      const int k0 = kc * 32;
      v16h afrag, bfrag;
#pragma unroll
      for (int j = 0; j < 16; ++j) {
        // ---- A gather (im2col on the fly, f16 activations) ----
        int kk = k0 + kbaseA + (j < 8 ? j : j + 8);
        _Float16 av = (_Float16)0.f;
        if (mvalid && kk < KTOT) {
          int cin = kk / (KS * KS * KS);
          int rk  = kk - cin * (KS * KS * KS);
          int kz  = rk / (KS * KS);
          int r2  = rk - kz * KS * KS;
          int ky  = r2 / KS;
          int kx  = r2 - ky * KS;
          int iz = oz * STR - PAD + kz * DIL;
          int iy = oy * STR - PAD + ky * DIL;
          int ix = ox * STR - PAD + kx * DIL;
          if ((unsigned)iz < (unsigned)ID && (unsigned)iy < (unsigned)ID &&
              (unsigned)ix < (unsigned)ID)
            av = xin[((size_t)cin * ID + iz) * ID * ID + (size_t)iy * ID + ix];
        }
        afrag[j] = av;
        // ---- B fragment from LDS-staged tile ----
        int kb = k0 + kbaseB + (j < 8 ? j : j + 8);
        _Float16 bv = (_Float16)0.f;
        if (kb < KTOT) bv = lds_w[mrow16 * KTOT + kb];
        bfrag[j] = bv;
      }
      acc = __builtin_amdgcn_wmma_f32_16x16x32_f16(
          /*neg_a=*/false, afrag, /*neg_b=*/false, bfrag,
          /*c_mod=*/(short)0, acc, /*reuse_a=*/false, /*reuse_b=*/false);
    }

    // fused epilogue: conv bias + BN + ReLU, store f16 (ISA C/D layout)
    const float sc = scale[ncol];
    const float sh = shift[ncol];
    _Float16* __restrict__ yp = y + (size_t)n * COUT * M + (size_t)ncol * M;
    const int mb = mt * 16 + ((lane < 16) ? 0 : 8);
    if (mb + 8 <= M) {                       // full tile: no per-element exec masking
#pragma unroll
      for (int r = 0; r < 8; ++r) {
        float v = fmaxf(acc[r] * sc + sh, 0.f);
        yp[mb + r] = (_Float16)v;
      }
    } else {                                 // tail tile only
#pragma unroll
      for (int r = 0; r < 8; ++r) {
        if (mb + r < M) {
          float v = fmaxf(acc[r] * sc + sh, 0.f);
          yp[mb + r] = (_Float16)v;
        }
      }
    }
  }
}

// ===================== attention merge =====================
__global__ void attention_kernel(const float* __restrict__ s8,
                                 const float* __restrict__ sb8,
                                 float* __restrict__ merged_ws,
                                 float* __restrict__ merged_out) {
  int b = blockIdx.x;
  int lane = threadIdx.x;  // 32 threads
  __shared__ float F[27][16];
  __shared__ float att[27];
  __shared__ float red[2];
  for (int t = lane; t < 27 * 16; t += 32) {
    int i = t / 16, f = t % 16;
    F[i][f] = (f < 8) ? s8[(b * 27 + i) * 8 + f] : sb8[(b * 27 + i) * 8 + (f - 8)];
  }
  __syncthreads();
  if (lane < 27) {
    float acc = 0.f;
    for (int i = 0; i < 27; ++i) {
      float d = 0.f;
      for (int f = 0; f < 16; ++f) d += F[i][f] * F[lane][f];
      acc += d;
    }
    att[lane] = acc / 27.f;
  }
  __syncthreads();
  if (lane == 0) {
    float mx = -3.0e38f;
    for (int j = 0; j < 27; ++j) mx = fmaxf(mx, att[j]);
    float sm = 0.f;
    for (int j = 0; j < 27; ++j) sm += expf(att[j] - mx);
    red[0] = mx; red[1] = sm;
  }
  __syncthreads();
  if (lane < 27) {
    float p = expf(att[lane] - red[0]) / red[1];
    for (int f = 0; f < 16; ++f) {
      float v = F[lane][f] * p;
      merged_ws[(b * 27 + lane) * 16 + f] = v;
      merged_out[(b * 27 + lane) * 16 + f] = v;
    }
  }
}

// h = [merged(432) | x2f(128)] -> fc3(560->128) -> fc4(128->8) -> fc5(8->1)
__global__ void final_fc_kernel(const float* __restrict__ merged_ws,
                                const float* __restrict__ x2f,
                                const float* __restrict__ w3, const float* __restrict__ b3,
                                const float* __restrict__ w4, const float* __restrict__ b4,
                                const float* __restrict__ w5, const float* __restrict__ b5,
                                float* __restrict__ out) {
  int b = blockIdx.x;
  int t = threadIdx.x;  // 128 threads
  __shared__ float h[128];
  __shared__ float h2[8];
  {
    const float* wp = w3 + (long)t * 560;
    const float* m = merged_ws + b * 432;
    const float* x = x2f + b * 128;
    float s = b3[t];
    for (int j = 0; j < 432; ++j) s += m[j] * wp[j];
    for (int j = 0; j < 128; ++j) s += x[j] * wp[432 + j];
    h[t] = s;
  }
  __syncthreads();
  if (t < 8) {
    const float* wp = w4 + (long)t * 128;
    float s = b4[t];
    for (int j = 0; j < 128; ++j) s += h[j] * wp[j];
    h2[t] = s;
  }
  __syncthreads();
  if (t == 0) {
    float s = b5[0];
    for (int j = 0; j < 8; ++j) s += h2[j] * w5[j];
    out[b] = s;
  }
}

// ===================== host side =====================

// d_in indices: setup_inputs() dict insertion order
enum {
  IX1 = 0, IX2, IX3,
  M_C1W, M_C1B, M_B1G, M_B1B, M_B1M, M_B1V,
  M_C2W, M_C2B, M_B2G, M_B2B, M_B2M, M_B2V,
  M_C3W, M_C3B, M_B3G, M_B3B, M_B3M, M_B3V,
  M_C4W, M_C4B, M_B4G, M_B4B, M_B4M, M_B4V,
  M_FCW, M_FCB,
  S_C1W, S_C1B, S_B1G, S_B1B, S_B1M, S_B1V,
  S_C2W, S_C2B, S_B2G, S_B2B, S_B2M, S_B2V,
  S_C3W, S_C3B, S_B3G, S_B3B, S_B3M, S_B3V,
  S_FC1W, S_FC1B, S_FC2W, S_FC2B,
  U_C1W, U_C1B, U_B1G, U_B1B, U_B1M, U_B1V,
  U_C2W, U_C2B, U_B2G, U_B2B, U_B2M, U_B2V,
  U_FC1W, U_FC1B, U_FC2W, U_FC2B,
  FSUBW, FSUBB, FC3W, FC3B, FC4W, FC4B, FC5W, FC5B
};

template <int CIN, int COUT, int KS, int STR, int PAD, int DIL, int ID, int OD>
static void launch_conv(const _Float16* x, const _Float16* w, const float* sc,
                        const float* sh, _Float16* y, int nsamp, hipStream_t stream) {
  constexpr int M = OD * OD * OD;
  int mtiles = (M + 15) / 16;
  dim3 grid((mtiles + 3) / 4, COUT / 16, nsamp);
  conv3d_wmma_kernel<CIN, COUT, KS, STR, PAD, DIL, ID, OD>
      <<<grid, 128, 0, stream>>>(x, w, sc, sh, y, mtiles);
}

extern "C" void kernel_launch(void* const* d_in, const int* in_sizes, int n_in,
                              void* d_out, int out_size, void* d_ws, size_t ws_size,
                              hipStream_t stream) {
  (void)in_sizes; (void)n_in; (void)out_size; (void)ws_size;
  const int B = 4, NS1 = 108, NSUB = 37044;
  auto F = [&](int i) { return (const float*)d_in[i]; };

  // deterministic linear allocator over d_ws
  char* base = (char*)d_ws;
  size_t off = 0;
  auto alloc = [&](size_t bytes) -> char* {
    char* p = base + off;
    off += (bytes + 255) & ~(size_t)255;
    return p;
  };

  // ping-pong half arenas: A holds conv outputs, Bar holds cast inputs / pool outputs
  _Float16* A   = (_Float16*)alloc((size_t)74088000 * 2);  // max: sub conv1 out
  _Float16* Bar = (_Float16*)alloc((size_t)37933056 * 2);  // max: sub pool1 out

  // f16 weight caches
  _Float16* w_m1 = (_Float16*)alloc(432 * 2);
  _Float16* w_m2 = (_Float16*)alloc(27648 * 2);
  _Float16* w_m3 = (_Float16*)alloc(221184 * 2);
  _Float16* w_m4 = (_Float16*)alloc(262144 * 2);
  _Float16* w_s1 = (_Float16*)alloc(432 * 2);
  _Float16* w_s2 = (_Float16*)alloc(13824 * 2);
  _Float16* w_s3 = (_Float16*)alloc(55296 * 2);
  _Float16* w_u1 = (_Float16*)alloc(128 * 2);
  _Float16* w_u2 = (_Float16*)alloc(4096 * 2);

  // fused BN scale/shift
  float* sc_m1 = (float*)alloc(16 * 4);  float* sh_m1 = (float*)alloc(16 * 4);
  float* sc_m2 = (float*)alloc(64 * 4);  float* sh_m2 = (float*)alloc(64 * 4);
  float* sc_m3 = (float*)alloc(128 * 4); float* sh_m3 = (float*)alloc(128 * 4);
  float* sc_m4 = (float*)alloc(256 * 4); float* sh_m4 = (float*)alloc(256 * 4);
  float* sc_s1 = (float*)alloc(16 * 4);  float* sh_s1 = (float*)alloc(16 * 4);
  float* sc_s2 = (float*)alloc(32 * 4);  float* sh_s2 = (float*)alloc(32 * 4);
  float* sc_s3 = (float*)alloc(64 * 4);  float* sh_s3 = (float*)alloc(64 * 4);
  float* sc_u1 = (float*)alloc(16 * 4);  float* sh_u1 = (float*)alloc(16 * 4);
  float* sc_u2 = (float*)alloc(32 * 4);  float* sh_u2 = (float*)alloc(32 * 4);

  // small f32 buffers
  float* sub_mean = (float*)alloc((size_t)NSUB * 32 * 4);
  float* sub_f1   = (float*)alloc((size_t)NSUB * 24 * 4);
  float* sb2      = (float*)alloc((size_t)NSUB * 2 * 4);   // == [108,686]
  float* sbfc     = (float*)alloc((size_t)NS1 * 8 * 4);
  float* sgl_mean = (float*)alloc((size_t)NS1 * 64 * 4);
  float* sgl_f1   = (float*)alloc((size_t)NS1 * 24 * 4);
  float* sbuf     = (float*)alloc((size_t)NS1 * 8 * 4);
  float* whl_mean = (float*)alloc((size_t)B * 256 * 4);
  float* x2f      = (float*)alloc((size_t)B * 128 * 4);
  float* merged_ws = (float*)alloc((size_t)B * 432 * 4);

  auto castk = [&](const float* src, _Float16* dst, long n) {
    cast_f32_f16_kernel<<<(int)((n + 255) / 256), 256, 0, stream>>>(src, dst, n);
  };
  auto bnp = [&](int gi, int cbi, float* sc, float* sh, int C) {
    bn_prep_kernel<<<(C + 255) / 256, 256, 0, stream>>>(
        F(gi), F(gi + 1), F(gi + 2), F(gi + 3), F(cbi), sc, sh, C);
  };
  auto pool = [&](const _Float16* x, _Float16* y, long NC, int ID, int OD, int k, int s) {
    long total = NC * (long)OD * OD * OD;
    maxpool3d_kernel<<<(int)((total + 255) / 256), 256, 0, stream>>>(x, y, total, ID, OD, k, s);
  };
  auto meanp = [&](const _Float16* x, float* y, long NC, int V) {
    meanpool_kernel<<<(int)((NC + 255) / 256), 256, 0, stream>>>(x, y, NC, V);
  };
  auto lin = [&](const float* x, int wi, float* y, int N, int IN, int OUT) {
    linear_kernel<<<((N * OUT) + 255) / 256, 256, 0, stream>>>(x, F(wi), F(wi + 1), y, N, IN, OUT);
  };

  // ---- weight casts + BN folding (tiny) ----
  castk(F(M_C1W), w_m1, 432);    castk(F(M_C2W), w_m2, 27648);
  castk(F(M_C3W), w_m3, 221184); castk(F(M_C4W), w_m4, 262144);
  castk(F(S_C1W), w_s1, 432);    castk(F(S_C2W), w_s2, 13824);
  castk(F(S_C3W), w_s3, 55296);
  castk(F(U_C1W), w_u1, 128);    castk(F(U_C2W), w_u2, 4096);
  bnp(M_B1G, M_C1B, sc_m1, sh_m1, 16);  bnp(M_B2G, M_C2B, sc_m2, sh_m2, 64);
  bnp(M_B3G, M_C3B, sc_m3, sh_m3, 128); bnp(M_B4G, M_C4B, sc_m4, sh_m4, 256);
  bnp(S_B1G, S_C1B, sc_s1, sh_s1, 16);  bnp(S_B2G, S_C2B, sc_s2, sh_s2, 32);
  bnp(S_B3G, S_C3B, sc_s3, sh_s3, 64);
  bnp(U_B1G, U_C1B, sc_u1, sh_u1, 16);  bnp(U_B2G, U_C2B, sc_u2, sh_u2, 32);

  // ---- sub branch: 37044 x [1,7,7,7] ----
  castk(F(IX3), Bar, (long)NSUB * 343);
  launch_conv<1, 16, 2, 2, 2, 2, 7, 5>(Bar, w_u1, sc_u1, sh_u1, A, NSUB, stream);
  pool(A, Bar, (long)NSUB * 16, 5, 4, 2, 1);
  launch_conv<16, 32, 2, 2, 2, 2, 4, 3>(Bar, w_u2, sc_u2, sh_u2, A, NSUB, stream);
  pool(A, Bar, (long)NSUB * 32, 3, 2, 2, 1);
  meanp(Bar, sub_mean, (long)NSUB * 32, 8);
  lin(sub_mean, U_FC1W, sub_f1, NSUB, 32, 24);
  lin(sub_f1, U_FC2W, sb2, NSUB, 24, 2);     // [37044,2] == [108,686]
  lin(sb2, FSUBW, sbfc, NS1, 686, 8);        // fc_sub -> [108,8]

  // ---- single branch: 108 x [1,48,48,48] ----
  castk(F(IX1), Bar, (long)NS1 * 110592);
  launch_conv<1, 16, 3, 2, 2, 2, 48, 24>(Bar, w_s1, sc_s1, sh_s1, A, NS1, stream);
  pool(A, Bar, (long)NS1 * 16, 24, 23, 2, 1);
  launch_conv<16, 32, 3, 2, 2, 2, 23, 12>(Bar, w_s2, sc_s2, sh_s2, A, NS1, stream);
  pool(A, Bar, (long)NS1 * 32, 12, 11, 2, 1);
  launch_conv<32, 64, 3, 2, 2, 2, 11, 6>(Bar, w_s3, sc_s3, sh_s3, A, NS1, stream);
  pool(A, Bar, (long)NS1 * 64, 6, 5, 2, 1);
  meanp(Bar, sgl_mean, (long)NS1 * 64, 125);
  lin(sgl_mean, S_FC1W, sgl_f1, NS1, 64, 24);
  lin(sgl_f1, S_FC2W, sbuf, NS1, 24, 8);     // s -> [108,8]

  // ---- whole-image branch: 4 x [1,96,96,96] ----
  castk(F(IX2), Bar, (long)B * 884736);
  launch_conv<1, 16, 3, 1, 0, 1, 96, 94>(Bar, w_m1, sc_m1, sh_m1, A, B, stream);
  pool(A, Bar, (long)B * 16, 94, 46, 3, 2);
  launch_conv<16, 64, 3, 1, 0, 2, 46, 42>(Bar, w_m2, sc_m2, sh_m2, A, B, stream);
  pool(A, Bar, (long)B * 64, 42, 20, 3, 2);
  launch_conv<64, 128, 3, 1, 2, 2, 20, 20>(Bar, w_m3, sc_m3, sh_m3, A, B, stream);
  pool(A, Bar, (long)B * 128, 20, 9, 3, 2);
  launch_conv<128, 256, 2, 1, 1, 2, 9, 9>(Bar, w_m4, sc_m4, sh_m4, A, B, stream);
  pool(A, Bar, (long)B * 256, 9, 3, 5, 2);
  meanp(Bar, whl_mean, (long)B * 256, 27);
  lin(whl_mean, M_FCW, x2f, B, 256, 128);    // x2f -> [4,128]

  // ---- attention merge + head ----
  float* out_f = (float*)d_out;
  attention_kernel<<<B, 32, 0, stream>>>(sbuf, sbfc, merged_ws, out_f + B);
  final_fc_kernel<<<B, 128, 0, stream>>>(merged_ws, x2f,
                                         F(FC3W), F(FC3B), F(FC4W), F(FC4B),
                                         F(FC5W), F(FC5B), out_f);
}
